// MixedAttentionBlock_59923383714549
// MI455X (gfx1250) — matrix operand order repaired and checked by reading, not compiled, not run on hardware
//
#include <hip/hip_runtime.h>
#include <hip/hip_bf16.h>

// ---------------------------------------------------------------------------
// Types for CDNA5 WMMA (gfx1250, wave32)
// ---------------------------------------------------------------------------
typedef __attribute__((ext_vector_type(16))) __bf16 v16bf;
typedef __attribute__((ext_vector_type(8)))  __bf16 v8bf;
typedef __attribute__((ext_vector_type(8)))  float  v8f;

__device__ inline v8f wmma_bf16(v16bf a, v16bf b, v8f c) {
  // D = A(16x32 bf16) x B(32x16 bf16) + C(16x16 f32)
  return __builtin_amdgcn_wmma_f32_16x16x32_bf16(
      /*neg_a=*/false, a, /*neg_b=*/false, b,
      /*c_mod=*/(short)0, c, /*reuse_a=*/false, /*reuse_b=*/false);
}

#define BB 4
#define TT 2048
#define DD 512
#define HH 8
#define HDD 64
#define FFD 2048
#define MM (BB * TT)        // 8192 rows
#define TDQ 1536            // 3*D

// ---------------------------------------------------------------------------
// Elementwise: f32 -> bf16
// ---------------------------------------------------------------------------
__global__ __launch_bounds__(256) void cvt_bf16(const float* __restrict__ x,
                                                __bf16* __restrict__ y, int n) {
  int i = blockIdx.x * 256 + threadIdx.x;
  if (i < n) y[i] = (__bf16)x[i];
}

// W (K x N, f32, row-major) -> Wt (N x K, bf16, row-major)  (i.e. B^T layout)
__global__ __launch_bounds__(256) void wtrans(const float* __restrict__ w,
                                              __bf16* __restrict__ wt,
                                              int K, int N) {
  int idx = blockIdx.x * 256 + threadIdx.x;
  if (idx >= K * N) return;
  int n = idx / K, k = idx % K;
  wt[idx] = (__bf16)w[(size_t)k * N + n];
}

// ---------------------------------------------------------------------------
// GEMM: C(MxN) = A(MxK bf16, row-major) @ Wt^T (Wt is NxK bf16) + bias
// block = 256 threads = 8 waves arranged 4(M) x 2(N); wave tile 32x64
// (2 M-tiles x 4 N-tiles => 8 WMMAs per 12 B128 loads per K-step).
// Block tile 128 x 128. Fragment loads follow CDNA5 16-bit A/B VGPR layouts.
// ---------------------------------------------------------------------------
__global__ __launch_bounds__(256) void gemm_bf16(const __bf16* __restrict__ A,
                                                 const __bf16* __restrict__ Wt,
                                                 const float* __restrict__ bias,
                                                 float* __restrict__ Cf,
                                                 __bf16* __restrict__ Cb,
                                                 int N, int K, int gelu) {
  const int w = threadIdx.x >> 5, lane = threadIdx.x & 31;
  const int wm = w & 3, wn = w >> 2;
  const int m0 = blockIdx.y * 128 + wm * 32;
  const int n0 = blockIdx.x * 128 + wn * 64;
  const int r = lane & 15, half = lane >> 4;

  v8f acc[2][4];
#pragma unroll
  for (int a = 0; a < 2; ++a)
#pragma unroll
    for (int b = 0; b < 4; ++b) acc[a][b] = (v8f){0, 0, 0, 0, 0, 0, 0, 0};

  for (int kk = 0; kk < K; kk += 32) {
    v16bf afr[2], bfr[4];
#pragma unroll
    for (int tm = 0; tm < 2; ++tm) {
      // A 16x32 layout: lanes<16 hold K {0..7,16..23}, lanes>=16 {8..15,24..31}
      const __bf16* pa = A + (size_t)(m0 + tm * 16 + r) * K + kk + half * 8;
      if (tm == 0) __builtin_prefetch(pa + 64, 0, 0);   // next-next K block
      v8bf lo = *(const v8bf*)pa;
      v8bf hi = *(const v8bf*)(pa + 16);
#pragma unroll
      for (int i = 0; i < 8; ++i) { afr[tm][i] = lo[i]; afr[tm][i + 8] = hi[i]; }
    }
#pragma unroll
    for (int tn = 0; tn < 4; ++tn) {
      // B 32x16 layout: lane = column n; lanes<16 hold K {0..7,16..23}, lanes>=16 {8..15,24..31}
      const __bf16* pb = Wt + (size_t)(n0 + tn * 16 + r) * K + kk + half * 8;
      if (tn == 0) __builtin_prefetch(pb + 64, 0, 0);
      v8bf lo = *(const v8bf*)pb;
      v8bf hi = *(const v8bf*)(pb + 16);
#pragma unroll
      for (int i = 0; i < 8; ++i) { bfr[tn][i] = lo[i]; bfr[tn][i + 8] = hi[i]; }
    }
#pragma unroll
    for (int tm = 0; tm < 2; ++tm)
#pragma unroll
      for (int tn = 0; tn < 4; ++tn)
        acc[tm][tn] = wmma_bf16(afr[tm], bfr[tn], acc[tm][tn]);
  }

  // Epilogue. C layout: lane holds col n=lane&15; VGPR v holds row v+8*(lane>>4)
#pragma unroll
  for (int tn = 0; tn < 4; ++tn) {
    float bv = bias[n0 + tn * 16 + r];
#pragma unroll
    for (int tm = 0; tm < 2; ++tm) {
#pragma unroll
      for (int v = 0; v < 8; ++v) {
        float val = acc[tm][tn][v] + bv;
        if (gelu) val = 0.5f * val * (1.0f + erff(val * 0.70710678118f));
        size_t off = (size_t)(m0 + tm * 16 + v + 8 * half) * N + n0 + tn * 16 + r;
        if (Cf) Cf[off] = val;
        if (Cb) Cb[off] = (__bf16)val;
      }
    }
  }
}

// ---------------------------------------------------------------------------
// Local windowed attention (WIN=8 => <=17 keys). One wave per (b,t,h).
// qkv is f32 (M x 1536): [q | k | v]. Writes bf16 output (M x 512).
// Streaming (online) softmax, wave32 shuffle reductions over HD=64 (2/lane).
// ---------------------------------------------------------------------------
__global__ __launch_bounds__(256) void local_attn(const float* __restrict__ qkv,
                                                  __bf16* __restrict__ out) {
  int w = threadIdx.x >> 5, lane = threadIdx.x & 31;
  int gw = blockIdx.x * 8 + w;                 // B*T*H = 65536 waves
  int h = gw & 7;
  int t = (gw >> 3) & (TT - 1);
  int bb = gw >> 14;
  const float* qp = qkv + (size_t)(bb * TT + t) * TDQ + h * HDD;
  float q0 = qp[lane] * 0.125f, q1 = qp[32 + lane] * 0.125f;
  int slo = (t > 8) ? t - 8 : 0;
  int shi = (t + 8 < TT - 1) ? t + 8 : TT - 1;
  float m = -1e30f, l = 0.f, o0 = 0.f, o1 = 0.f;
  for (int s = slo; s <= shi; ++s) {
    const float* kp = qkv + (size_t)(bb * TT + s) * TDQ + DD + h * HDD;
    float d = q0 * kp[lane] + q1 * kp[32 + lane];
#pragma unroll
    for (int off = 16; off; off >>= 1) d += __shfl_xor(d, off, 32);
    float mn = fmaxf(m, d);
    float al = __expf(m - mn);
    float p = __expf(d - mn);
    const float* vp = qkv + (size_t)(bb * TT + s) * TDQ + 2 * DD + h * HDD;
    o0 = o0 * al + p * vp[lane];
    o1 = o1 * al + p * vp[32 + lane];
    l = l * al + p;
    m = mn;
  }
  __bf16* op = out + (size_t)(bb * TT + t) * DD + h * HDD;
  op[lane] = (__bf16)(o0 / l);
  op[32 + lane] = (__bf16)(o1 / l);
}

// ---------------------------------------------------------------------------
// Global MHA, flash-style. One wave per 16-query tile of one (b,h).
// qkv bf16 (M x 1536). Output g bf16 (M x 512). 32-key blocks, online softmax.
// S = Q K^T via WMMA (2 k-steps over HD=64); O += P V via WMMA, P transposed
// through LDS, V gathered via LDS into B-layout.
// ---------------------------------------------------------------------------
__global__ __launch_bounds__(128) void flash_mha(const __bf16* __restrict__ qkv,
                                                 __bf16* __restrict__ g) {
  __shared__ __align__(16) float  pst[4][16][32];
  __shared__ __align__(16) __bf16 vlds[4][32][64];
  const int w = threadIdx.x >> 5, lane = threadIdx.x & 31;
  const int r = lane & 15, half = lane >> 4;
  int gw = blockIdx.x * 4 + w;           // 4096 waves total
  int qb = gw & 127;                     // T/16 = 128 query tiles
  int bh = gw >> 7;
  int bb = bh >> 3, h = bh & 7;
  int t0 = qb * 16;

  // Load resident Q fragments (16 x 64 = two 16x32 A-fragments)
  v16bf qa[2];
  {
    const __bf16* qp = qkv + (size_t)(bb * TT + t0 + r) * TDQ + h * HDD;
#pragma unroll
    for (int ks = 0; ks < 2; ++ks) {
      v8bf lo = *(const v8bf*)(qp + ks * 32 + half * 8);
      v8bf hi = *(const v8bf*)(qp + ks * 32 + 16 + half * 8);
#pragma unroll
      for (int i = 0; i < 8; ++i) { qa[ks][i] = lo[i]; qa[ks][i + 8] = hi[i]; }
    }
  }
  v8f o[4];
#pragma unroll
  for (int t = 0; t < 4; ++t) o[t] = (v8f){0, 0, 0, 0, 0, 0, 0, 0};
  float mrow[8], lrow[8];
#pragma unroll
  for (int v = 0; v < 8; ++v) { mrow[v] = -1e30f; lrow[v] = 0.f; }

  for (int j = 0; j < TT / 32; ++j) {
    // Stage V block (32 keys x 64 dims) into LDS: lane loads one key row.
    {
      const __bf16* vp = qkv + (size_t)(bb * TT + j * 32 + lane) * TDQ + 2 * DD + h * HDD;
      v8bf* dst = (v8bf*)&vlds[w][lane][0];
#pragma unroll
      for (int i = 0; i < 8; ++i) dst[i] = *(const v8bf*)(vp + i * 8);
    }
    // S tiles: keys [j*32, j*32+16) and [j*32+16, j*32+32)
    v8f s0 = (v8f){0, 0, 0, 0, 0, 0, 0, 0}, s1 = s0;
#pragma unroll
    for (int ks = 0; ks < 2; ++ks) {
      const __bf16* kp0 = qkv + (size_t)(bb * TT + j * 32 + r) * TDQ + DD + h * HDD + ks * 32 + half * 8;
      const __bf16* kp1 = qkv + (size_t)(bb * TT + j * 32 + 16 + r) * TDQ + DD + h * HDD + ks * 32 + half * 8;
      v16bf kb0, kb1;
      v8bf l0 = *(const v8bf*)kp0, h0 = *(const v8bf*)(kp0 + 16);
      v8bf l1 = *(const v8bf*)kp1, h1 = *(const v8bf*)(kp1 + 16);
#pragma unroll
      for (int i = 0; i < 8; ++i) {
        kb0[i] = l0[i]; kb0[i + 8] = h0[i];
        kb1[i] = l1[i]; kb1[i + 8] = h1[i];
      }
      s0 = wmma_bf16(qa[ks], kb0, s0);
      s1 = wmma_bf16(qa[ks], kb1, s1);
    }
    // Online softmax update; C-layout row = v + 8*half, col = r.
#pragma unroll
    for (int v = 0; v < 8; ++v) {
      float a0 = s0[v] * 0.125f, a1 = s1[v] * 0.125f;
      float mx = fmaxf(a0, a1);
#pragma unroll
      for (int off = 8; off; off >>= 1) mx = fmaxf(mx, __shfl_xor(mx, off, 32));
      float mnew = fmaxf(mrow[v], mx);
      float alpha = __expf(mrow[v] - mnew);
      mrow[v] = mnew;
      float p0 = __expf(a0 - mnew), p1 = __expf(a1 - mnew);
      pst[w][v + 8 * half][r] = p0;
      pst[w][v + 8 * half][16 + r] = p1;
      float rs = p0 + p1;
#pragma unroll
      for (int off = 8; off; off >>= 1) rs += __shfl_xor(rs, off, 32);
      lrow[v] = lrow[v] * alpha + rs;
      o[0][v] *= alpha; o[1][v] *= alpha; o[2][v] *= alpha; o[3][v] *= alpha;
    }
    __syncthreads();
    // P tile -> A-fragment layout (lane = row, packed K = key index)
    v16bf pa;
#pragma unroll
    for (int i = 0; i < 8; ++i) {
      pa[i]     = (__bf16)pst[w][r][half * 16 + i];
      pa[i + 8] = (__bf16)pst[w][r][half * 16 + 8 + i];
    }
    // V -> B-fragment layout via LDS gather; O += P V
#pragma unroll
    for (int tn = 0; tn < 4; ++tn) {
      v16bf vb;
#pragma unroll
      for (int jj = 0; jj < 8; ++jj) {
        int k0 = half * 16 + 2 * jj;
        vb[2 * jj]     = vlds[w][k0][tn * 16 + r];
        vb[2 * jj + 1] = vlds[w][k0 + 1][tn * 16 + r];
      }
      o[tn] = wmma_bf16(pa, vb, o[tn]);
    }
    __syncthreads();
  }
  // Normalize and store bf16
#pragma unroll
  for (int tn = 0; tn < 4; ++tn) {
#pragma unroll
    for (int v = 0; v < 8; ++v) {
      float val = o[tn][v] / lrow[v];
      int row = v + 8 * half;
      g[(size_t)(bb * TT + t0 + row) * DD + h * HDD + tn * 16 + r] = (__bf16)val;
    }
  }
}

// ---------------------------------------------------------------------------
// Residual add + LayerNorm per row (D=512). 256 threads, 2 elems/thread.
// ---------------------------------------------------------------------------
__device__ inline float block_reduce_sum(float v, float* red) {
  __syncthreads();
  int lane = threadIdx.x & 31, w = threadIdx.x >> 5;
#pragma unroll
  for (int off = 16; off; off >>= 1) v += __shfl_xor(v, off, 32);
  if (lane == 0) red[w] = v;
  __syncthreads();
  float t = (threadIdx.x < 8) ? red[threadIdx.x] : 0.f;
#pragma unroll
  for (int off = 4; off; off >>= 1) t += __shfl_xor(t, off, 32);
  if (threadIdx.x == 0) red[0] = t;
  __syncthreads();
  return red[0];
}

__global__ __launch_bounds__(256) void add_ln(const float* __restrict__ a,
                                              const float* __restrict__ c,
                                              const float* __restrict__ s,
                                              const float* __restrict__ bi,
                                              float* __restrict__ outf,
                                              __bf16* __restrict__ outb) {
  __shared__ float red[8];
  size_t row = blockIdx.x;
  const float* pa = a + row * DD;
  const float* pc = c + row * DD;
  int i0 = threadIdx.x, i1 = threadIdx.x + 256;
  float v0 = pa[i0] + pc[i0], v1 = pa[i1] + pc[i1];
  float mean = block_reduce_sum(v0 + v1, red) * (1.f / DD);
  float d0 = v0 - mean, d1 = v1 - mean;
  float var = block_reduce_sum(d0 * d0 + d1 * d1, red) * (1.f / DD);
  float inv = rsqrtf(var + 1e-5f);
  float o0 = d0 * inv * s[i0] + bi[i0];
  float o1 = d1 * inv * s[i1] + bi[i1];
  float* po = outf + row * DD;
  po[i0] = o0; po[i1] = o1;
  if (outb) {
    __bf16* pb = outb + row * DD;
    pb[i0] = (__bf16)o0; pb[i1] = (__bf16)o1;
  }
}

// ---------------------------------------------------------------------------
// SE path (tiny): mean over T, two small dense layers, channel scale.
// ---------------------------------------------------------------------------
__global__ __launch_bounds__(256) void mean_t(const float* __restrict__ x,
                                              float* __restrict__ s) {
  int idx = blockIdx.x * 256 + threadIdx.x;  // B*D = 2048
  if (idx >= BB * DD) return;
  int bb = idx >> 9, d = idx & (DD - 1);
  float acc = 0.f;
  for (int t = 0; t < TT; ++t) acc += x[(size_t)(bb * TT + t) * DD + d];
  s[idx] = acc * (1.f / TT);
}

__global__ __launch_bounds__(256) void se_fc1(const float* __restrict__ s,
                                              const float* __restrict__ w,
                                              const float* __restrict__ b,
                                              float* __restrict__ out) {
  int idx = blockIdx.x * 256 + threadIdx.x;  // B*RED = 256
  if (idx >= BB * 64) return;
  int bb = idx >> 6, r = idx & 63;
  float acc = b[r];
  for (int d = 0; d < DD; ++d) acc += s[bb * DD + d] * w[(size_t)d * 64 + r];
  out[idx] = fmaxf(acc, 0.f);
}

__global__ __launch_bounds__(256) void se_fc2(const float* __restrict__ s1,
                                              const float* __restrict__ w,
                                              const float* __restrict__ b,
                                              float* __restrict__ out) {
  int idx = blockIdx.x * 256 + threadIdx.x;  // B*D = 2048
  if (idx >= BB * DD) return;
  int bb = idx >> 9, d = idx & (DD - 1);
  float acc = b[d];
  for (int r = 0; r < 64; ++r) acc += s1[bb * 64 + r] * w[(size_t)r * DD + d];
  out[idx] = 1.f / (1.f + __expf(-acc));
}

__global__ __launch_bounds__(256) void se_scale(const float* __restrict__ x,
                                                const float* __restrict__ s,
                                                float* __restrict__ outf,
                                                __bf16* __restrict__ outb) {
  size_t i = (size_t)blockIdx.x * 256 + threadIdx.x;  // B*T*D
  if (i >= (size_t)MM * DD) return;
  int bb = (int)(i >> 20);                 // / (T*D) = / 1048576
  int d = (int)(i & (DD - 1));
  float v = x[i] * (1.f + s[bb * DD + d]);
  outf[i] = v;
  outb[i] = (__bf16)v;
}

// ---------------------------------------------------------------------------
// Host orchestration
// ---------------------------------------------------------------------------
extern "C" void kernel_launch(void* const* d_in, const int* in_sizes, int n_in,
                              void* d_out, int out_size, void* d_ws, size_t ws_size,
                              hipStream_t stream) {
  (void)in_sizes; (void)n_in; (void)out_size; (void)ws_size;
  const float* x       = (const float*)d_in[0];
  const float* qkv_w   = (const float*)d_in[1];
  const float* qkv_b   = (const float*)d_in[2];
  const float* proj_w  = (const float*)d_in[3];
  const float* proj_b  = (const float*)d_in[4];
  const float* ln1_s   = (const float*)d_in[5];
  const float* ln1_b   = (const float*)d_in[6];
  const float* min_w   = (const float*)d_in[7];
  const float* min_b   = (const float*)d_in[8];
  const float* mout_w  = (const float*)d_in[9];
  const float* mout_b  = (const float*)d_in[10];
  const float* ln2_s   = (const float*)d_in[11];
  const float* ln2_b   = (const float*)d_in[12];
  const float* se1_w   = (const float*)d_in[13];
  const float* se1_b   = (const float*)d_in[14];
  const float* se2_w   = (const float*)d_in[15];
  const float* se2_b   = (const float*)d_in[16];
  const float* ff1_w   = (const float*)d_in[17];
  const float* ff1_b   = (const float*)d_in[18];
  const float* ff2_w   = (const float*)d_in[19];
  const float* ff2_b   = (const float*)d_in[20];
  const float* ln3_s   = (const float*)d_in[21];
  const float* ln3_b   = (const float*)d_in[22];
  float* out = (float*)d_out;

  size_t off = 0;
  auto arena = [&](size_t bytes) -> char* {
    char* p = (char*)d_ws + off;
    off += (bytes + 255) & ~(size_t)255;
    return p;
  };
  __bf16* wt_qkv  = (__bf16*)arena((size_t)1536 * 512 * 2);
  __bf16* wt_proj = (__bf16*)arena((size_t)512 * 512 * 2);
  __bf16* wt_min  = (__bf16*)arena((size_t)1536 * 512 * 2);
  __bf16* wt_mout = (__bf16*)arena((size_t)512 * 512 * 2);
  __bf16* wt_ff1  = (__bf16*)arena((size_t)2048 * 512 * 2);
  __bf16* wt_ff2  = (__bf16*)arena((size_t)512 * 2048 * 2);
  __bf16* act16   = (__bf16*)arena((size_t)MM * DD * 2);      // xb/ab/x1b/gb/x3b
  char*   qkvreg  = arena((size_t)MM * TDQ * 4);              // qkv f32 | qkv bf16
  float*  qkv_f   = (float*)qkvreg;
  __bf16* qkvb    = (__bf16*)qkvreg;
  float*  bufY    = (float*)arena((size_t)MM * DD * 4);       // transient GEMM out
  float*  x1f     = (float*)arena((size_t)MM * DD * 4);
  float*  x2f     = (float*)arena((size_t)MM * DD * 4);
  float*  x3f     = (float*)arena((size_t)MM * DD * 4);
  __bf16* hb      = (__bf16*)arena((size_t)MM * FFD * 2);
  float*  sb      = (float*)arena((size_t)BB * DD * 4);
  float*  s1b     = (float*)arena((size_t)BB * 64 * 4);
  float*  s2b     = (float*)arena((size_t)BB * DD * 4);

  // Weight transposes -> bf16 (N x K)
  wtrans<<<(512 * 1536 + 255) / 256, 256, 0, stream>>>(qkv_w,  wt_qkv,  512, 1536);
  wtrans<<<(512 * 512  + 255) / 256, 256, 0, stream>>>(proj_w, wt_proj, 512, 512);
  wtrans<<<(512 * 1536 + 255) / 256, 256, 0, stream>>>(min_w,  wt_min,  512, 1536);
  wtrans<<<(512 * 512  + 255) / 256, 256, 0, stream>>>(mout_w, wt_mout, 512, 512);
  wtrans<<<(512 * 2048 + 255) / 256, 256, 0, stream>>>(ff1_w,  wt_ff1,  512, 2048);
  wtrans<<<(2048 * 512 + 255) / 256, 256, 0, stream>>>(ff2_w,  wt_ff2,  2048, 512);

  // (1) local windowed attention + residual + LN
  cvt_bf16<<<(MM * DD) / 256, 256, 0, stream>>>(x, act16, MM * DD);
  gemm_bf16<<<dim3(1536 / 128, MM / 128), 256, 0, stream>>>(
      act16, wt_qkv, qkv_b, qkv_f, nullptr, 1536, 512, 0);
  local_attn<<<(MM * HH) / 8, 256, 0, stream>>>(qkv_f, act16);
  gemm_bf16<<<dim3(512 / 128, MM / 128), 256, 0, stream>>>(
      act16, wt_proj, proj_b, bufY, nullptr, 512, 512, 0);
  add_ln<<<MM, 256, 0, stream>>>(x, bufY, ln1_s, ln1_b, x1f, act16);

  // (2) global MHA + residual + LN
  gemm_bf16<<<dim3(1536 / 128, MM / 128), 256, 0, stream>>>(
      act16, wt_min, min_b, nullptr, qkvb, 1536, 512, 0);
  flash_mha<<<(BB * HH * (TT / 16)) / 4, 128, 0, stream>>>(qkvb, act16);
  gemm_bf16<<<dim3(512 / 128, MM / 128), 256, 0, stream>>>(
      act16, wt_mout, mout_b, bufY, nullptr, 512, 512, 0);
  add_ln<<<MM, 256, 0, stream>>>(x1f, bufY, ln2_s, ln2_b, x2f, nullptr);

  // (3) channel SE, additive residual
  mean_t<<<(BB * DD + 255) / 256, 256, 0, stream>>>(x2f, sb);
  se_fc1<<<1, 256, 0, stream>>>(sb, se1_w, se1_b, s1b);
  se_fc2<<<(BB * DD + 255) / 256, 256, 0, stream>>>(s1b, se2_w, se2_b, s2b);
  se_scale<<<(MM * DD) / 256, 256, 0, stream>>>(x2f, s2b, x3f, act16);

  // (4) FFN (exact GELU) + residual + LN
  gemm_bf16<<<dim3(2048 / 128, MM / 128), 256, 0, stream>>>(
      act16, wt_ff1, ff1_b, nullptr, hb, 2048, 512, 1);
  gemm_bf16<<<dim3(512 / 128, MM / 128), 256, 0, stream>>>(
      hb, wt_ff2, ff2_b, bufY, nullptr, 512, 2048, 0);
  add_ln<<<MM, 256, 0, stream>>>(x3f, bufY, ln3_s, ln3_b, out, nullptr);
}
